// SamAutoMasker_88837103551010
// MI455X (gfx1250) — compile-verified, matrix operand order checked by reading, not run on hardware
//
#include <hip/hip_runtime.h>
#include <hip/hip_bf16.h>

typedef __attribute__((ext_vector_type(16))) _Float16 v16h;
typedef __attribute__((ext_vector_type(8)))  float    v8f;
typedef __attribute__((ext_vector_type(4)))  float    v4f;

#define N_MASKS 1024
#define IMG_H 256
#define IMG_W 256
#define HW 65536
#define NHW ((size_t)N_MASKS * (size_t)HW)

#define PRED_IOU_THRESH 0.88f
#define STABILITY_THRESH 0.95f
#define BOX_NMS_THRESH 0.7f

__device__ __forceinline__ int imin(int a, int b) { return a < b ? a : b; }
__device__ __forceinline__ int imax(int a, int b) { return a > b ? a : b; }

__device__ __forceinline__ float box_iou(v4f a, v4f b) {
    float x0 = fmaxf(a[0], b[0]);
    float y0 = fmaxf(a[1], b[1]);
    float x1 = fminf(a[2], b[2]);
    float y1 = fminf(a[3], b[3]);
    float inter = fmaxf(x1 - x0, 0.f) * fmaxf(y1 - y0, 0.f);
    float aa = fmaxf(a[2] - a[0], 0.f) * fmaxf(a[3] - a[1], 0.f);
    float ab = fmaxf(b[2] - b[0], 0.f) * fmaxf(b[3] - b[1], 0.f);
    return inter / fmaxf(aa + ab - inter, 1e-6f);
}

// ---------------------------------------------------------------------------
// Pass 1: per-mask stability counts (WMMA ones-matmul) + bounding boxes.
// One workgroup (8 wave32s) per mask; each wave covers 16 chunks of 512
// elements. Counting runs on the matrix pipes (v_wmma_f32_16x16x32_f16 vs an
// all-ones B; lane-sum of C == 16 * count, exact in f32). Box extents are
// fully branch-free: a 16-bit predicate mask per lane + ffs/clz.
// Loads use the default temporal hint so the tensor tail stays resident in
// the 192 MB L2 for pass 3's reversed re-read.
// ---------------------------------------------------------------------------
__global__ __launch_bounds__(256)
void sam_stats_kernel(const float* __restrict__ logits,
                      const float* __restrict__ iou_preds,
                      float* __restrict__ ws_boxes,   // [N,4]
                      float* __restrict__ ws_score,   // [N]
                      float* __restrict__ ws_valid)   // [N]
{
    const int n    = blockIdx.x;
    const int lane = threadIdx.x & 31;
    const int wave = threadIdx.x >> 5;
    const float* base = logits + (size_t)n * HW;

    v16h ones;
#pragma unroll
    for (int j = 0; j < 16; ++j) ones[j] = (_Float16)1.0f;

    v8f c_hi = {};
    v8f c_lo = {};

    int xmn = IMG_W, xmx = -1, ymn = IMG_H, ymx = -1;

#pragma unroll 2
    for (int it = 0; it < 16; ++it) {
        const int chunk = wave + (it << 3);           // 0..127
        const int off   = (chunk << 9) + (lane << 4); // element offset in mask
        const float* p  = base + off;

        // Branch-free prefetch of the next chunk (clamped inside this mask).
        __builtin_prefetch(base + imin(off + (8 << 9), HW - 16), 0, 1);

        v4f q0 = *((const v4f*)p + 0);
        v4f q1 = *((const v4f*)p + 1);
        v4f q2 = *((const v4f*)p + 2);
        v4f q3 = *((const v4f*)p + 3);

        float vals[16];
#pragma unroll
        for (int j = 0; j < 4; ++j) {
            vals[j] = q0[j]; vals[4 + j] = q1[j];
            vals[8 + j] = q2[j]; vals[12 + j] = q3[j];
        }

        const int y  = off >> 8;          // all 16 elems are in this row
        const int x0 = off & (IMG_W - 1); // first column of the lane's span

        v16h a_hi, a_lo;
        unsigned pm = 0u;                 // bit j <=> vals[j] > 0
#pragma unroll
        for (int j = 0; j < 16; ++j) {
            float v = vals[j];
            a_hi[j] = (v >  1.0f) ? (_Float16)1.0f : (_Float16)0.0f;
            a_lo[j] = (v > -1.0f) ? (_Float16)1.0f : (_Float16)0.0f;
            pm |= (v > 0.0f) ? (1u << j) : 0u;
        }

        // Branch-free box extents: min/max set bit of pm -> column range.
        const bool any = (pm != 0u);
        const int  flo = __ffs(pm) - 1;       // garbage when !any (unselected)
        const int  fhi = 31 - __clz(pm);
        xmn = imin(xmn, any ? (x0 + flo) : IMG_W);
        xmx = imax(xmx, any ? (x0 + fhi) : -1);
        ymn = imin(ymn, any ? y : IMG_H);
        ymx = imax(ymx, any ? y : -1);

        // D = A * ones + C : XDL ops co-execute with the VALU work above.
        c_hi = __builtin_amdgcn_wmma_f32_16x16x32_f16(
            false, a_hi, false, ones, (short)0, c_hi, false, false);
        c_lo = __builtin_amdgcn_wmma_f32_16x16x32_f16(
            false, a_lo, false, ones, (short)0, c_lo, false, false);
    }

    float hs = 0.f, ls = 0.f;
#pragma unroll
    for (int j = 0; j < 8; ++j) { hs += c_hi[j]; ls += c_lo[j]; }

    // wave32 cross-lane reduction
#pragma unroll
    for (int m = 16; m >= 1; m >>= 1) {
        hs += __shfl_xor(hs, m, 32);
        ls += __shfl_xor(ls, m, 32);
        xmn = imin(xmn, __shfl_xor(xmn, m, 32));
        xmx = imax(xmx, __shfl_xor(xmx, m, 32));
        ymn = imin(ymn, __shfl_xor(ymn, m, 32));
        ymx = imax(ymx, __shfl_xor(ymx, m, 32));
    }

    __shared__ float s_h[8], s_l[8];
    __shared__ int   s_xmn[8], s_xmx[8], s_ymn[8], s_ymx[8];
    if (lane == 0) {
        s_h[wave] = hs;   s_l[wave] = ls;
        s_xmn[wave] = xmn; s_xmx[wave] = xmx;
        s_ymn[wave] = ymn; s_ymx[wave] = ymx;
    }
    __syncthreads();

    if (threadIdx.x == 0) {
        float hsum = 0.f, lsum = 0.f;
        int a = IMG_W, b = -1, c = IMG_H, d = -1;
#pragma unroll
        for (int w = 0; w < 8; ++w) {
            hsum += s_h[w]; lsum += s_l[w];
            a = imin(a, s_xmn[w]); b = imax(b, s_xmx[w]);
            c = imin(c, s_ymn[w]); d = imax(d, s_ymx[w]);
        }
        float hi = hsum * (1.0f / 16.0f);   // undo the 16x N-replication
        float lo = lsum * (1.0f / 16.0f);
        float stab = hi / fmaxf(lo, 1.0f);
        float iou  = iou_preds[n];
        float valid = ((iou > PRED_IOU_THRESH) && (stab >= STABILITY_THRESH))
                          ? 1.0f : 0.0f;
        bool empty = (b < 0);
        v4f box;
        box[0] = empty ? 0.f : (float)a;
        box[1] = empty ? 0.f : (float)c;
        box[2] = empty ? 0.f : (float)b;
        box[3] = empty ? 0.f : (float)d;
        *((v4f*)ws_boxes + n) = box;
        ws_score[n] = iou;
        ws_valid[n] = valid;
    }
}

// ---------------------------------------------------------------------------
// Pass 2: greedy NMS, single 1024-thread workgroup, all state in LDS (~30 KB
// of the 320 KB WGP pool). Argmax-select / keep / suppress; equivalent to the
// reference's stable-sorted fori_loop (strict '>' keeps lowest idx on ties).
// ---------------------------------------------------------------------------
__global__ __launch_bounds__(1024)
void sam_nms_kernel(const float* __restrict__ ws_boxes,
                    const float* __restrict__ ws_score,
                    const float* __restrict__ ws_valid,
                    float* __restrict__ ws_gate,
                    float* __restrict__ out_tail)  // d_out + N*H*W
{
    const int t = threadIdx.x;
    __shared__ v4f           s_box[N_MASKS];
    __shared__ float         s_score[N_MASKS];
    __shared__ float         s_val[N_MASKS];
    __shared__ int           s_idx[N_MASKS];
    __shared__ unsigned char s_active[N_MASKS];
    __shared__ unsigned char s_keep[N_MASKS];

    s_box[t]    = *((const v4f*)ws_boxes + t);
    s_score[t]  = ws_score[t];
    s_active[t] = (ws_valid[t] > 0.5f) ? 1 : 0;
    s_keep[t]   = 0;
    __syncthreads();

    for (int iter = 0; iter < N_MASKS; ++iter) {
        s_val[t] = s_active[t] ? s_score[t] : -1e30f;
        s_idx[t] = t;
        __syncthreads();
        for (int s = 512; s > 0; s >>= 1) {
            if (t < s) {
                if (s_val[t + s] > s_val[t]) {
                    s_val[t] = s_val[t + s];
                    s_idx[t] = s_idx[t + s];
                }
            }
            __syncthreads();
        }
        const float best_v = s_val[0];
        const int   best   = s_idx[0];
        __syncthreads();
        if (best_v < -1e29f) break;           // uniform across the workgroup

        v4f bb = s_box[best];
        if (t == best) {
            s_keep[t] = 1;
            s_active[t] = 0;
        } else if (s_active[t]) {
            if (box_iou(bb, s_box[t]) > BOX_NMS_THRESH) s_active[t] = 0;
        }
        __syncthreads();
    }

    float k = s_keep[t] ? 1.0f : 0.0f;
    ws_gate[t]  = k * s_score[t];
    out_tail[t] = k;                                   // keep mask (as f32)
    ((v4f*)(out_tail + N_MASKS))[t] = s_box[t];        // boxes
}

// ---------------------------------------------------------------------------
// Pass 3: out = sigmoid(logits) * gate[n]. Streaming 256 MiB in + 256 MiB
// out. Masks are processed in REVERSE order so the logits tail still
// resident in the 192 MB L2 from pass 1 is re-read before eviction.
// Stores are NT (output never re-read -> don't evict the logits).
// ---------------------------------------------------------------------------
__global__ __launch_bounds__(256)
void sam_output_kernel(const float* __restrict__ logits,
                       const float* __restrict__ ws_gate,
                       float* __restrict__ out)
{
    const int n = (N_MASKS - 1) - blockIdx.x;   // reverse for L2 reuse
    const float g = ws_gate[n];
    const v4f* in4 = (const v4f*)(logits + (size_t)n * HW);
    v4f*       o4  = (v4f*)(out + (size_t)n * HW);
    const float NLOG2E = -1.4426950408889634f;

    for (int i = threadIdx.x; i < HW / 4; i += 256) {
        v4f v = in4[i];
        v4f r;
#pragma unroll
        for (int j = 0; j < 4; ++j) {
            float e = __builtin_amdgcn_exp2f(v[j] * NLOG2E);
            r[j] = g * __builtin_amdgcn_rcpf(1.0f + e);
        }
        __builtin_nontemporal_store(r, o4 + i);
    }
}

extern "C" void kernel_launch(void* const* d_in, const int* in_sizes, int n_in,
                              void* d_out, int out_size, void* d_ws, size_t ws_size,
                              hipStream_t stream) {
    const float* logits = (const float*)d_in[0];   // [1024,256,256] f32
    const float* ious   = (const float*)d_in[1];   // [1024] f32
    float* out = (float*)d_out;                    // [1024*256*256 + 1024 + 4096]
    float* ws  = (float*)d_ws;

    float* ws_boxes = ws;          // 4096 floats
    float* ws_score = ws + 4096;   // 1024 floats
    float* ws_valid = ws + 5120;   // 1024 floats
    float* ws_gate  = ws + 6144;   // 1024 floats

    sam_stats_kernel<<<N_MASKS, 256, 0, stream>>>(logits, ious,
                                                  ws_boxes, ws_score, ws_valid);
    sam_nms_kernel<<<1, 1024, 0, stream>>>(ws_boxes, ws_score, ws_valid,
                                           ws_gate, out + NHW);
    sam_output_kernel<<<N_MASKS, 256, 0, stream>>>(logits, ws_gate, out);
}
